// Net_25469156065564
// MI455X (gfx1250) — matrix operand order repaired and checked by reading, not compiled, not run on hardware
//
#include <hip/hip_runtime.h>
#include <hip/hip_bf16.h>

// ---------------------------------------------------------------------------
// Types for CDNA5 WMMA (wave32): v_wmma_f32_16x16x32_bf16
// ---------------------------------------------------------------------------
typedef __bf16 bf16;
typedef __attribute__((ext_vector_type(16))) __bf16 v16bf;
typedef __attribute__((ext_vector_type(8)))  __bf16 v8bf;
typedef __attribute__((ext_vector_type(8)))  float  v8f;

// ---------------------------------------------------------------------------
// Stage 1: fused embedding gather -> Tpad[b, lpad(104), 768] bf16
//   k in [0,250)  : word_emb[token]
//   k in [250,375): pos1_emb[pos1]
//   k in [375,500): pos2_emb[pos2]
//   k in [500,750): sdp * word_emb[token]
//   k in [750,768): 0 (K padding)   lpad 0,1,102,103 : 0 (L padding)
// ---------------------------------------------------------------------------
__global__ __launch_bounds__(256)
void build_tpad(const int* __restrict__ tok, const int* __restrict__ p1,
                const int* __restrict__ p2, const float* __restrict__ sdp,
                const float* __restrict__ wemb, const float* __restrict__ p1e,
                const float* __restrict__ p2e, bf16* __restrict__ T)
{
    int bi = blockIdx.x;              // 0 .. 512*104-1
    int b  = bi / 104, lp = bi % 104;
    long base = (long)bi * 768;
    int  l = lp - 2;
    bool pad = (l < 0) || (l >= 100);
    int t = 0, q1 = 0, q2 = 0; float s = 0.f;
    if (!pad) {
        int idx = b * 100 + l;
        t = tok[idx]; q1 = p1[idx]; q2 = p2[idx]; s = sdp[idx];
    }
    for (int k = threadIdx.x; k < 768; k += blockDim.x) {
        float v = 0.f;
        if (!pad) {
            if      (k < 250) v = wemb[(long)t * 250 + k];
            else if (k < 375) v = p1e[q1 * 125 + (k - 250)];
            else if (k < 500) v = p2e[q2 * 125 + (k - 375)];
            else if (k < 750) v = s * wemb[(long)t * 250 + (k - 500)];
        }
        T[base + k] = (bf16)v;
    }
}

// ---------------------------------------------------------------------------
// Weight repacking to bf16 GEMM layout [Npad][Ktot], zero padded.
// conv1 (OIHW (100,250,3,kw)) -> W[o][dw*768 + h*250 + c], Npad=128
// ---------------------------------------------------------------------------
__global__ void prep_w1(const float* __restrict__ src, bf16* __restrict__ dst, int kw)
{
    int  Kp = kw * 768;
    long total = (long)128 * Kp;
    for (long i = (long)blockIdx.x * blockDim.x + threadIdx.x; i < total;
         i += (long)gridDim.x * blockDim.x) {
        int o = (int)(i / Kp), rem = (int)(i % Kp);
        int dw = rem / 768, kp = rem % 768;
        float v = 0.f;
        if (o < 100 && kp < 750) {
            int h = kp / 250, c = kp % 250;
            v = src[((o * 250 + c) * 3 + h) * kw + dw];
        }
        dst[i] = (bf16)v;
    }
}

// conv2a (200,300,1,1) -> W[o][k], Npad=224, Ktot=320
__global__ void prep_w2a(const float* __restrict__ src, bf16* __restrict__ dst)
{
    long total = 224L * 320;
    for (long i = (long)blockIdx.x * blockDim.x + threadIdx.x; i < total;
         i += (long)gridDim.x * blockDim.x) {
        int o = (int)(i / 320), k = (int)(i % 320);
        dst[i] = (bf16)((o < 200 && k < 300) ? src[o * 300 + k] : 0.f);
    }
}

// conv2b (300,200,1,3) -> W[o][dw*224 + c], Npad=320, Ktot=672
__global__ void prep_w2b(const float* __restrict__ src, bf16* __restrict__ dst)
{
    long total = 320L * 672;
    for (long i = (long)blockIdx.x * blockDim.x + threadIdx.x; i < total;
         i += (long)gridDim.x * blockDim.x) {
        int o = (int)(i / 672), k = (int)(i % 672);
        int dw = k / 224, c = k % 224;
        dst[i] = (bf16)((o < 300 && c < 200) ? src[(o * 200 + c) * 3 + dw] : 0.f);
    }
}

// ---------------------------------------------------------------------------
// Generic implicit-GEMM, WMMA bf16 -> f32, 32x32 tile per wave (register
// blocked 2x2), fused bias (+ optional ReLU).
//   C[row, col] = sum_k A[base(row) + k] * W[col*Ktot + k]
//   base(row)   = ((row/divq)*pitch + row%divq + off) * lda    (contiguous K)
// Fragment layouts per CDNA5 ISA 7.12.2:
//   A (16x32 bf16): lane m=l&15, half=l>>4; e0..7 -> K=half*8+e,
//                   e8..15 -> K=16+half*8+(e-8)
//   B (32x16 bf16): lane n=l&15, half=l>>4; e0..7 -> K=half*8+e,
//                   e8..15 -> K=16+half*8+(e-8)
//   C (16x16 f32) : lane n=l&15; vgpr r -> M = r + 8*half
// Per K-chunk: 8 b128 loads feed 4 WMMAs (2 loads/WMMA).
// All lanes active through the WMMA loop (wave-uniform tail guard only).
// ---------------------------------------------------------------------------
#define SHUF16(lo, hi) __builtin_shufflevector(lo, hi, \
    0,1,2,3,4,5,6,7,8,9,10,11,12,13,14,15)

__global__ __launch_bounds__(256)
void gemm_bf16(const bf16* __restrict__ A, const bf16* __restrict__ W,
               float* __restrict__ O, const float* __restrict__ bias,
               int Mreal, int Nreal, int Ktot, int Ntiles2,
               int lda, int divq, int pitch, int off,
               int ldO, int ncolOff, int doRelu)
{
    int lane = threadIdx.x & 31;
    int wave = threadIdx.x >> 5;
    int Mtiles2 = (Mreal + 31) >> 5;
    long t = (long)blockIdx.x * 8 + wave;
    if (t >= (long)Mtiles2 * Ntiles2) return;    // wave-uniform: EXEC all-1s
    int mt = (int)(t / Ntiles2);
    int nt = (int)(t % Ntiles2);

    int half = lane >> 4;
    int mn   = lane & 15;

    int r0 = mt * 32 + mn;       if (r0 >= Mreal) r0 = Mreal - 1;
    int r1 = mt * 32 + 16 + mn;  if (r1 >= Mreal) r1 = Mreal - 1;
    long baseA0 = ((long)(r0 / divq) * pitch + r0 % divq + off) * lda;
    long baseA1 = ((long)(r1 / divq) * pitch + r1 % divq + off) * lda;
    long baseB0 = (long)(nt * 32 + mn) * Ktot;
    long baseB1 = (long)(nt * 32 + 16 + mn) * Ktot;

    v8f acc00 = {0.f,0.f,0.f,0.f,0.f,0.f,0.f,0.f};
    v8f acc01 = acc00, acc10 = acc00, acc11 = acc00;

    for (int k0 = 0; k0 < Ktot; k0 += 32) {
        v8bf a0lo = *(const v8bf*)(A + baseA0 + k0 + half * 8);
        v8bf a0hi = *(const v8bf*)(A + baseA0 + k0 + 16 + half * 8);
        v8bf a1lo = *(const v8bf*)(A + baseA1 + k0 + half * 8);
        v8bf a1hi = *(const v8bf*)(A + baseA1 + k0 + 16 + half * 8);
        v8bf b0lo = *(const v8bf*)(W + baseB0 + k0 + half * 8);
        v8bf b0hi = *(const v8bf*)(W + baseB0 + k0 + 16 + half * 8);
        v8bf b1lo = *(const v8bf*)(W + baseB1 + k0 + half * 8);
        v8bf b1hi = *(const v8bf*)(W + baseB1 + k0 + 16 + half * 8);
        v16bf a0 = SHUF16(a0lo, a0hi);
        v16bf a1 = SHUF16(a1lo, a1hi);
        v16bf b0 = SHUF16(b0lo, b0hi);
        v16bf b1 = SHUF16(b1lo, b1hi);
        acc00 = __builtin_amdgcn_wmma_f32_16x16x32_bf16(
                    false, a0, false, b0, (short)0, acc00, false, false);
        acc01 = __builtin_amdgcn_wmma_f32_16x16x32_bf16(
                    false, a0, false, b1, (short)0, acc01, false, false);
        acc10 = __builtin_amdgcn_wmma_f32_16x16x32_bf16(
                    false, a1, false, b0, (short)0, acc10, false, false);
        acc11 = __builtin_amdgcn_wmma_f32_16x16x32_bf16(
                    false, a1, false, b1, (short)0, acc11, false, false);
    }

    // Epilogue: bias (+ReLU), masked store of the four 16x16 sub-tiles.
    #pragma unroll
    for (int sub = 0; sub < 4; ++sub) {
        const v8f& acc = (sub == 0) ? acc00 : (sub == 1) ? acc01
                       : (sub == 2) ? acc10 : acc11;
        int colBase = nt * 32 + ((sub & 1) ? 16 : 0);
        int rowBase = mt * 32 + ((sub & 2) ? 16 : 0);
        int col = colBase + mn;
        if (col < Nreal) {
            float bb = bias[col];
            #pragma unroll
            for (int r = 0; r < 8; ++r) {
                int row = rowBase + r + half * 8;
                if (row < Mreal) {
                    float v = acc[r] + bb;
                    if (doRelu) v = v > 0.f ? v : 0.f;
                    O[(long)row * ldO + ncolOff + col] = v;
                }
            }
        }
    }
}

// ---------------------------------------------------------------------------
// ReLU already fused in GEMM; pool width-2 over L, emit bf16 [B*50][320]
// ---------------------------------------------------------------------------
__global__ void pool1(const float* __restrict__ O1, bf16* __restrict__ P1)
{
    int r = blockIdx.x;                    // 0..25599 = b*50 + j
    int b = r / 50, j = r % 50;
    const float* a0 = O1 + (long)(b * 100 + 2 * j) * 300;
    const float* a1 = a0 + 300;
    bf16* out = P1 + (long)r * 320;
    for (int k = threadIdx.x; k < 320; k += blockDim.x) {
        float v = 0.f;
        if (k < 300) { float x = a0[k], y = a1[k]; v = x > y ? x : y; }
        out[k] = (bf16)v;
    }
}

// conv2a f32 [B*50][200] -> bf16 padded [B][52][224] (L-pad rows 0/51, C-pad)
__global__ void pad2a(const float* __restrict__ O2a, bf16* __restrict__ P2a)
{
    int bi = blockIdx.x;                   // 0..512*52-1
    int b = bi / 52, lp = bi % 52;
    bf16* out = P2a + (long)bi * 224;
    int  l = lp - 1;
    bool pad = (l < 0) || (l >= 50);
    const float* src = O2a + (long)(b * 50 + (pad ? 0 : l)) * 200;
    for (int k = threadIdx.x; k < 224; k += blockDim.x)
        out[k] = (bf16)((!pad && k < 200) ? src[k] : 0.f);
}

// Global max over the 50 positions (inputs are post-ReLU, >= 0)
__global__ void reduce_max(const float* __restrict__ O2b, float* __restrict__ F)
{
    int b = blockIdx.x;
    for (int c = threadIdx.x; c < 300; c += blockDim.x) {
        float m = 0.f;
        const float* p = O2b + (long)b * 50 * 300 + c;
        for (int l = 0; l < 50; ++l) m = fmaxf(m, p[l * 300]);
        F[b * 300 + c] = m;
    }
}

// Tiny MLP head: 300 ->relu 100 ->relu 50 -> 15   (f32, one block per sample)
__global__ __launch_bounds__(128)
void dense_head(const float* __restrict__ F,
                const float* __restrict__ d1w, const float* __restrict__ d1b,
                const float* __restrict__ d2w, const float* __restrict__ d2b,
                const float* __restrict__ d3w, const float* __restrict__ d3b,
                float* __restrict__ out)
{
    __shared__ float f[300], h1[100], h2[50];
    int b = blockIdx.x;
    for (int i = threadIdx.x; i < 300; i += 128) f[i] = F[b * 300 + i];
    __syncthreads();
    if (threadIdx.x < 100) {
        float s = d1b[threadIdx.x];
        for (int k = 0; k < 300; ++k) s += f[k] * d1w[k * 100 + threadIdx.x];
        h1[threadIdx.x] = s > 0.f ? s : 0.f;
    }
    __syncthreads();
    if (threadIdx.x < 50) {
        float s = d2b[threadIdx.x];
        for (int k = 0; k < 100; ++k) s += h1[k] * d2w[k * 50 + threadIdx.x];
        h2[threadIdx.x] = s > 0.f ? s : 0.f;
    }
    __syncthreads();
    if (threadIdx.x < 15) {
        float s = d3b[threadIdx.x];
        for (int k = 0; k < 50; ++k) s += h2[k] * d3w[k * 15 + threadIdx.x];
        out[b * 15 + threadIdx.x] = s;
    }
}

// ---------------------------------------------------------------------------
extern "C" void kernel_launch(void* const* d_in, const int* in_sizes, int n_in,
                              void* d_out, int out_size, void* d_ws, size_t ws_size,
                              hipStream_t stream)
{
    (void)in_sizes; (void)n_in; (void)out_size; (void)ws_size;
    const int*   tok  = (const int*)  d_in[0];
    const int*   p1   = (const int*)  d_in[1];
    const int*   p2   = (const int*)  d_in[2];
    const float* sdp  = (const float*)d_in[3];
    const float* wemb = (const float*)d_in[4];
    const float* p1e  = (const float*)d_in[5];
    const float* p2e  = (const float*)d_in[6];
    const float* w11  = (const float*)d_in[7];   const float* b11 = (const float*)d_in[8];
    const float* w13  = (const float*)d_in[9];   const float* b13 = (const float*)d_in[10];
    const float* w15  = (const float*)d_in[11];  const float* b15 = (const float*)d_in[12];
    const float* w2a  = (const float*)d_in[13];  const float* b2a = (const float*)d_in[14];
    const float* w2b  = (const float*)d_in[15];  const float* b2b = (const float*)d_in[16];
    const float* d1w  = (const float*)d_in[17];  const float* d1b = (const float*)d_in[18];
    const float* d2w  = (const float*)d_in[19];  const float* d2b = (const float*)d_in[20];
    const float* d3w  = (const float*)d_in[21];  const float* d3b = (const float*)d_in[22];
    float* out = (float*)d_out;

    char* ws = (char*)d_ws;
    size_t cur = 0;
    auto alloc = [&](size_t bytes) { size_t r = cur; cur += (bytes + 255) & ~(size_t)255; return r; };

    size_t oTPAD = alloc((size_t)512 * 104 * 768 * 2);   // 81.8 MB; reused for O2B
    size_t oO2B  = oTPAD;                                // 25600*300*4 = 30.7 MB fits
    size_t oO1   = alloc((size_t)51200 * 300 * 4);       // 61.4 MB; reused for O2A/P2A
    size_t oO2A  = oO1;                                  // 25600*200*4 = 20.5 MB
    size_t oP2A  = oO1 + (size_t)25600 * 200 * 4;        // 512*52*224*2 = 11.9 MB
    size_t oP1   = alloc((size_t)25600 * 320 * 2);       // 16.4 MB
    size_t oW11  = alloc((size_t)128 * 768  * 2);
    size_t oW13  = alloc((size_t)128 * 2304 * 2);
    size_t oW15  = alloc((size_t)128 * 3840 * 2);
    size_t oW2A  = alloc((size_t)224 * 320  * 2);
    size_t oW2B  = alloc((size_t)320 * 672  * 2);
    size_t oF    = alloc((size_t)512 * 300  * 4);

    bf16*  Tpad = (bf16*) (ws + oTPAD);
    float* O1   = (float*)(ws + oO1);
    bf16*  P1   = (bf16*) (ws + oP1);
    float* O2A  = (float*)(ws + oO2A);
    bf16*  P2A  = (bf16*) (ws + oP2A);
    float* O2B  = (float*)(ws + oO2B);
    float* Fbuf = (float*)(ws + oF);
    bf16 *W11 = (bf16*)(ws + oW11), *W13 = (bf16*)(ws + oW13), *W15 = (bf16*)(ws + oW15);
    bf16 *W2A = (bf16*)(ws + oW2A), *W2B = (bf16*)(ws + oW2B);

    // Stage 1: fused embedding gather (bf16, padded)
    build_tpad<<<512 * 104, 256, 0, stream>>>(tok, p1, p2, sdp, wemb, p1e, p2e, Tpad);

    // Stage 2: weight repack (tiny)
    prep_w1 <<<256, 256, 0, stream>>>(w11, W11, 1);
    prep_w1 <<<256, 256, 0, stream>>>(w13, W13, 3);
    prep_w1 <<<256, 256, 0, stream>>>(w15, W15, 5);
    prep_w2a<<<128, 256, 0, stream>>>(w2a, W2A);
    prep_w2b<<<128, 256, 0, stream>>>(w2b, W2B);

    // Stage 3: conv1 branches as WMMA implicit GEMM (M=51200, N=100, 32x32 tiles)
    {
        int Mt = 1600, Nt = 4, grid = (Mt * Nt + 7) / 8;
        gemm_bf16<<<grid, 256, 0, stream>>>(Tpad, W11, O1, b11,
            51200, 100,  768, Nt, 768, 100, 104, 2, 300,   0, 1);
        gemm_bf16<<<grid, 256, 0, stream>>>(Tpad, W13, O1, b13,
            51200, 100, 2304, Nt, 768, 100, 104, 1, 300, 100, 1);
        gemm_bf16<<<grid, 256, 0, stream>>>(Tpad, W15, O1, b15,
            51200, 100, 3840, Nt, 768, 100, 104, 0, 300, 200, 1);
    }

    // Stage 4: width-2 max pool -> bf16 (Tpad now dead)
    pool1<<<25600, 320, 0, stream>>>(O1, P1);

    // Stage 5: conv2a 1x1 GEMM (M=25600, N=200, K=320), no ReLU (O1 now dead)
    {
        int Mt = 800, Nt = 7, grid = (Mt * Nt + 7) / 8;
        gemm_bf16<<<grid, 256, 0, stream>>>(P1, W2A, O2A, b2a,
            25600, 200, 320, Nt, 320, 50, 50, 0, 200, 0, 0);
    }

    // Stage 6: pad + bf16 for 1x3 conv
    pad2a<<<512 * 52, 224, 0, stream>>>(O2A, P2A);

    // Stage 7: conv2b 1x3 GEMM (M=25600, N=300, K=672) + ReLU
    {
        int Mt = 800, Nt = 10, grid = (Mt * Nt + 7) / 8;
        gemm_bf16<<<grid, 256, 0, stream>>>(P2A, W2B, O2B, b2b,
            25600, 300, 672, Nt, 224, 50, 52, 0, 300, 0, 1);
    }

    // Stage 8: global max over 50 positions
    reduce_max<<<512, 256, 0, stream>>>(O2B, Fbuf);

    // Stage 9: MLP head
    dense_head<<<512, 128, 0, stream>>>(Fbuf, d1w, d1b, d2w, d2b, d3w, d3b, out);
}
